// VarianceAdaptor_60327110639729
// MI455X (gfx1250) — compile-verified, matrix operand order checked by reading, not run on hardware
//
#include <hip/hip_runtime.h>
#include <stdint.h>

// ---------------- problem constants ----------------
#define B_  32
#define S_  1024
#define T_  4096
#define D_  256
#define F_  256
#define NB_ 256

typedef __attribute__((ext_vector_type(16))) __bf16 v16bf;
typedef __attribute__((ext_vector_type(8)))  float  v8f;
typedef unsigned int v4u __attribute__((ext_vector_type(4)));
typedef int          v8i __attribute__((ext_vector_type(8)));
typedef int          v4i __attribute__((ext_vector_type(4)));

union Frag { v16bf v; uint4 q[2]; };

__device__ __forceinline__ unsigned short f32_to_bf16(float f) {
  unsigned u = __builtin_bit_cast(unsigned, f);
  unsigned r = u + 0x7FFFu + ((u >> 16) & 1u);   // round-to-nearest-even
  return (unsigned short)(r >> 16);
}
__device__ __forceinline__ float bf16_to_f32(unsigned short h) {
  unsigned u = ((unsigned)h) << 16;
  return __builtin_bit_cast(float, u);
}

// ---------------- weight pre-swizzle to WMMA B-fragment layout ----------------
// B operand of v_wmma_f32_16x16x32_bf16 is 32(K) x 16(N).
// Assumed lane striping (mirror of documented A layout):
//   lane = n + 16*half ;  q0 dwords j=0..3 hold K = 8*half + 2j,2j+1
//                         q1 dwords j=4..7 hold K = 16 + 8*half + 2(j-4),+1
// Packed per-fragment: lane's 16 bf16 stored contiguously (32B/lane, 1KB/frag).
__global__ void pack_weights_kernel(const float* w0, const float* w1, const float* w2,
                                    const float* w3, const float* w4, const float* w5,
                                    unsigned short* __restrict__ wpack) {
  int gid = blockIdx.x * blockDim.x + threadIdx.x;
  const int per_layer = 3 * 256 * 256;
  if (gid >= 6 * per_layer) return;
  int layer = gid / per_layer;
  int rem   = gid % per_layer;
  int k = rem / (256 * 256);
  int cf = rem % (256 * 256);
  int c = cf >> 8;          // Cin  (K-dim of GEMM)
  int f = cf & 255;         // Cout (N-dim of GEMM)
  const float* W = (layer == 0) ? w0 : (layer == 1) ? w1 : (layer == 2) ? w2
                 : (layer == 3) ? w3 : (layer == 4) ? w4 : w5;
  float val = W[((size_t)k * 256 + c) * 256 + f];

  int cc = c >> 5, r = c & 31;          // 32-wide K chunk, K within chunk
  int nb = f >> 4, n = f & 15;          // 16-wide N block, N within block
  int half, j;
  if (r < 16) { half = r >> 3;        j =     (r & 7) >> 1; }
  else        { half = (r - 16) >> 3; j = 4 + (((r - 16) & 7) >> 1); }
  int pos  = r & 1;
  int lane = n + 16 * half;
  int frag = (k * 8 + cc) * 16 + nb;    // 3*8*16 = 384 fragments per layer
  wpack[(size_t)layer * per_layer + (size_t)frag * 512 + lane * 16 + j * 2 + pos]
      = f32_to_bf16(val);
}

// ---------------- f32 -> bf16 activation convert ----------------
__global__ void convert_x_kernel(const float* __restrict__ x,
                                 unsigned short* __restrict__ xb, int n) {
  int gid = blockIdx.x * blockDim.x + threadIdx.x;
  if (gid < n) xb[gid] = f32_to_bf16(x[gid]);
}

// ---------------- per-batch inclusive scan of durations ----------------
__global__ void cumsum_kernel(const int* __restrict__ dur,
                              int* __restrict__ cum, int* __restrict__ mel_len) {
  __shared__ int sc[S_];
  int b = blockIdx.x, t = threadIdx.x;
  sc[t] = dur[b * S_ + t];
  __syncthreads();
  for (int off = 1; off < S_; off <<= 1) {
    int a  = sc[t];
    int ad = (t >= off) ? sc[t - off] : 0;
    __syncthreads();
    sc[t] = a + ad;
    __syncthreads();
  }
  cum[b * S_ + t] = sc[t];
  if (t == S_ - 1) mel_len[b] = min(sc[t], T_);
}

// ---------------- length-regulator index (searchsorted right) ----------------
__global__ void lr_index_kernel(const int* __restrict__ cum, int* __restrict__ idx) {
  int gid = blockIdx.x * blockDim.x + threadIdx.x;
  if (gid >= B_ * T_) return;
  int b = gid / T_, t = gid % T_;
  const int* c = cum + b * S_;
  int lo = 0, hi = S_;
  while (lo < hi) { int mid = (lo + hi) >> 1; if (c[mid] <= t) lo = mid + 1; else hi = mid; }
  idx[gid] = min(lo, S_ - 1);
}

// ---------------- gather expanded activations as bf16 ----------------
__global__ __launch_bounds__(256)
void gather_xr_kernel(const float* __restrict__ x, const int* __restrict__ idx,
                      const int* __restrict__ mel_len, unsigned short* __restrict__ xrb) {
  int bt = blockIdx.x;          // B*T blocks
  int d  = threadIdx.x;         // 256
  int b = bt / T_, t = bt % T_;
  int g = idx[bt];
  float v = (t < mel_len[b]) ? x[((size_t)b * S_ + g) * D_ + d] : 0.f;
  xrb[(size_t)bt * D_ + d] = f32_to_bf16(v);
}

// ---------------- WMMA conv(K=3,pad=1) + bias + ReLU + LayerNorm ----------------
#define AROWS 18
#define ASTR  264   // bf16 elems; 528B row stride -> conflict-free ds_load_b128
#define HSTR  264   // f32 elems

__global__ __launch_bounds__(256)
void conv_ln_kernel(const unsigned short* __restrict__ in,     // [B,L,256] bf16
                    const unsigned short* __restrict__ wpack,  // packed layer weights
                    const float* __restrict__ bias,
                    const float* __restrict__ gamma,
                    const float* __restrict__ beta,
                    unsigned short* __restrict__ out,          // [B,L,256] bf16
                    int L) {
  __shared__ unsigned short sA[AROWS * ASTR];
  __shared__ float sH[16 * HSTR];
  __shared__ float psum[16 * 16], psum2[16 * 16];
  __shared__ float smu[16], srs[16];

  int tiles = L >> 4;
  int b    = blockIdx.x / tiles;
  int tile = blockIdx.x % tiles;
  int t0   = tile << 4;
  int tid  = threadIdx.x;
  int wave = tid >> 5, lane = tid & 31;

  const uint4* inq = (const uint4*)(in + (size_t)b * L * 256);

#if __has_builtin(__builtin_amdgcn_tensor_load_to_lds)
  // ---- TDM: bulk-load interior rows t0..t0+15 into LDS rows 1..16 ----
  // 2D tensor: (B_*L) rows x 256 bf16; tile 16 x 256; LDS row stride 528B via
  // pad_enable (pad_interval=6 -> every 512B, pad_amount=3 -> +16B).
  if (wave == 0) {
    typedef __attribute__((address_space(3))) unsigned short lds_us;
    unsigned lds_base = (unsigned)(size_t)(lds_us*)(&sA[ASTR]);   // row 1
    unsigned long long gaddr =
        (unsigned long long)(size_t)in + ((unsigned long long)(b * L + t0) * 256ull) * 2ull;
    unsigned rows = (unsigned)(B_) * (unsigned)L;
    v4u g0;
    g0.x = 1u;                                            // count=1 (user descriptor)
    g0.y = lds_base;                                      // lds_addr
    g0.z = (unsigned)(gaddr & 0xFFFFFFFFull);             // global_addr[31:0]
    g0.w = (unsigned)((gaddr >> 32) & 0x01FFFFFFull)      // global_addr[56:32]
         | (2u << 30);                                    // type=2 ("image")
    v8i g1;
    g1[0] = (int)((1u << 16)        // data_size = 2 bytes
                | (1u << 20)        // pad_enable
                | (6u << 22)        // pad_interval: 128 dwords (512B)
                | (3u << 25));      // pad_amount:   4 dwords (16B)
    g1[1] = (int)(256u << 16);                              // tensor_dim0 = 256
    g1[2] = (int)((rows & 0xFFFFu) << 16);                  // tensor_dim1 lo16
    g1[3] = (int)(((rows >> 16) & 0xFFFFu) | (256u << 16)); // dim1 hi16 | tile_dim0=256
    g1[4] = 16;                                             // tile_dim1 = 16 rows
    g1[5] = 256;                                            // tensor_dim0_stride = 256
    g1[6] = 0; g1[7] = 0;
    v4i g2 = {0, 0, 0, 0};
    v4i g3 = {0, 0, 0, 0};
#if defined(__clang_major__) && (__clang_major__ >= 23)
    v8i g4 = {0, 0, 0, 0, 0, 0, 0, 0};
    __builtin_amdgcn_tensor_load_to_lds(g0, g1, g2, g3, g4, 0);
#else
    __builtin_amdgcn_tensor_load_to_lds(g0, g1, g2, g3, 0);
#endif
  }
  // halo rows (t0-1 -> LDS row 0, t0+16 -> LDS row 17), per-batch zero padding
  if (tid < 64) {
    int hrow = (tid >> 5) ? 17 : 0;
    int qi = tid & 31;
    int g = t0 - 1 + hrow;
    uint4 val = make_uint4(0u, 0u, 0u, 0u);
    if (g >= 0 && g < L) val = inq[(size_t)g * 32 + qi];
    *(uint4*)(&sA[hrow * ASTR + qi * 8]) = val;
  }
  if (wave == 0) {
#if __has_builtin(__builtin_amdgcn_s_wait_tensorcnt)
    __builtin_amdgcn_s_wait_tensorcnt(0);
#else
    asm volatile("s_wait_tensorcnt 0x0" ::: "memory");
#endif
  }
  __syncthreads();
#else
  // ---- fallback: stage whole halo tile (rows t0-1 .. t0+16) via VMEM ----
  for (int i = tid; i < AROWS * 32; i += 256) {
    int row = i >> 5, qi = i & 31;
    int g = t0 - 1 + row;
    uint4 val = make_uint4(0u, 0u, 0u, 0u);
    if (g >= 0 && g < L) val = inq[(size_t)g * 32 + qi];
    *(uint4*)(&sA[row * ASTR + qi * 8]) = val;
  }
  __syncthreads();
#endif

  // ---- 48 WMMAs per wave: 3 taps x 8 K-chunks x 2 N-blocks ----
  v8f acc0 = {0.f, 0.f, 0.f, 0.f, 0.f, 0.f, 0.f, 0.f};
  v8f acc1 = {0.f, 0.f, 0.f, 0.f, 0.f, 0.f, 0.f, 0.f};
  int m = lane & 15, half = lane >> 4;
  for (int k = 0; k < 3; ++k) {
    const unsigned short* arow = &sA[(m + k) * ASTR];
    #pragma unroll
    for (int cc = 0; cc < 8; ++cc) {
      int c = cc << 5;
      Frag fa, fb0, fb1;
      const unsigned short* ap = arow + c + 8 * half;
      fa.q[0] = *(const uint4*)(ap);
      fa.q[1] = *(const uint4*)(ap + 16);
      const uint4* bp = (const uint4*)(wpack + (size_t)((k * 8 + cc) * 16 + wave * 2) * 512)
                        + lane * 2;
      fb0.q[0] = bp[0];  fb0.q[1] = bp[1];
      fb1.q[0] = bp[64]; fb1.q[1] = bp[65];
      acc0 = __builtin_amdgcn_wmma_f32_16x16x32_bf16(false, fa.v, false, fb0.v,
                                                     (short)0, acc0, false, false);
      acc1 = __builtin_amdgcn_wmma_f32_16x16x32_bf16(false, fa.v, false, fb1.v,
                                                     (short)0, acc1, false, false);
    }
  }

  // ---- bias + ReLU, spill tile to LDS (C layout: vgpr v -> row v+8*half) ----
  int nA = (wave << 5) + (lane & 15);
  int nB = nA + 16;
  float biasA = bias[nA], biasB = bias[nB];
  #pragma unroll
  for (int v = 0; v < 8; ++v) {
    int row = v + 8 * half;
    float h0 = acc0[v] + biasA; h0 = h0 > 0.f ? h0 : 0.f;
    float h1 = acc1[v] + biasB; h1 = h1 > 0.f ? h1 : 0.f;
    sH[row * HSTR + nA] = h0;
    sH[row * HSTR + nB] = h1;
  }
  __syncthreads();

  // ---- LayerNorm over 256 channels per row ----
  int r = tid >> 4, seg = tid & 15;
  float s = 0.f, s2 = 0.f;
  #pragma unroll
  for (int i = 0; i < 16; ++i) {
    float hv = sH[r * HSTR + (seg << 4) + i];
    s += hv; s2 += hv * hv;
  }
  psum[r * 16 + seg] = s; psum2[r * 16 + seg] = s2;
  __syncthreads();
  if (tid < 16) {
    float ts = 0.f, ts2 = 0.f;
    #pragma unroll
    for (int i = 0; i < 16; ++i) { ts += psum[tid * 16 + i]; ts2 += psum2[tid * 16 + i]; }
    float mu  = ts * (1.f / 256.f);
    float var = ts2 * (1.f / 256.f) - mu * mu;
    smu[tid] = mu;
    srs[tid] = rsqrtf(var + 1e-5f);
  }
  __syncthreads();
  float mu = smu[r], rs = srs[r];
  size_t obase = ((size_t)b * L + t0 + r) * 256;
  #pragma unroll
  for (int i = 0; i < 16; ++i) {
    int cI = (seg << 4) + i;
    float v = (sH[r * HSTR + cI] - mu) * rs * gamma[cI] + beta[cI];
    out[obase + cI] = f32_to_bf16(v);
  }
}

// ---------------- final 256->1 linear (+ optional mel mask) ----------------
__global__ __launch_bounds__(256)
void linear_kernel(const unsigned short* __restrict__ h, const float* __restrict__ lw,
                   const float* __restrict__ lbp, const int* __restrict__ mel_len,
                   float* __restrict__ outp, int L, int maskMode) {
  int wave = threadIdx.x >> 5, lane = threadIdx.x & 31;
  int pos = blockIdx.x * 8 + wave;       // < B_*L
  int b = pos / L, t = pos % L;
  const unsigned short* hp = h + (size_t)pos * 256;
  float s = 0.f;
  #pragma unroll
  for (int i = 0; i < 8; ++i) {
    int c = lane + (i << 5);
    s += bf16_to_f32(hp[c]) * lw[c];
  }
  for (int off = 16; off > 0; off >>= 1) s += __shfl_down(s, off, 32);
  if (lane == 0) {
    float o = s + lbp[0];
    if (maskMode && t >= mel_len[b]) o = 0.f;
    outp[pos] = o;
  }
}

// ---------------- out = xr + pitch_emb + energy_emb ----------------
__global__ __launch_bounds__(256)
void final_out_kernel(const float* __restrict__ x, const int* __restrict__ idx,
                      const int* __restrict__ mel_len,
                      const float* __restrict__ ptarget, const float* __restrict__ etarget,
                      const float* __restrict__ pbins, const float* __restrict__ ebins,
                      const float* __restrict__ pemb, const float* __restrict__ eemb,
                      float* __restrict__ outp) {
  int bt = blockIdx.x;
  int d  = threadIdx.x;
  int b = bt / T_, t = bt % T_;
  float pv = ptarget[bt], ev = etarget[bt];
  // searchsorted(side='left') over 255 bins -> bucket in [0,255]
  int lo = 0, hi = NB_ - 1;
  while (lo < hi) { int mid = (lo + hi) >> 1; if (pbins[mid] < pv) lo = mid + 1; else hi = mid; }
  int pb = lo;
  lo = 0; hi = NB_ - 1;
  while (lo < hi) { int mid = (lo + hi) >> 1; if (ebins[mid] < ev) lo = mid + 1; else hi = mid; }
  int eb = lo;
  int g = idx[bt];
  float xv = (t < mel_len[b]) ? x[((size_t)b * S_ + g) * D_ + d] : 0.f;
  outp[(size_t)bt * D_ + d] = xv + pemb[(size_t)pb * D_ + d] + eemb[(size_t)eb * D_ + d];
}

// ---------------- mel_len (as float) + mel_mask outputs ----------------
__global__ void mask_len_kernel(const int* __restrict__ mel_len,
                                float* __restrict__ out_len, float* __restrict__ out_mask) {
  int gid = blockIdx.x * blockDim.x + threadIdx.x;
  if (gid < B_) out_len[gid] = (float)mel_len[gid];
  if (gid < B_ * T_) {
    int b = gid / T_, t = gid % T_;
    out_mask[gid] = (t >= mel_len[b]) ? 1.f : 0.f;
  }
}

// ---------------- host launcher ----------------
extern "C" void kernel_launch(void* const* d_in, const int* in_sizes, int n_in,
                              void* d_out, int out_size, void* d_ws, size_t ws_size,
                              hipStream_t stream) {
  // input order (setup_inputs insertion order, params flattened depth-first):
  // 0:x 1:src_mask 2:mel_mask 3:duration_target 4:pitch_target 5:energy_target 6:max_len
  // 7..16  dur:    w1 b1 g1 bt1 w2 b2 g2 bt2 lw lb
  // 17..26 pitch:  same
  // 27..36 energy: same
  // 37:pitch_bins 38:energy_bins 39:pitch_emb 40:energy_emb
  const float* x   = (const float*)d_in[0];
  const int*   dtg = (const int*)d_in[3];
  const float* ptg = (const float*)d_in[4];
  const float* etg = (const float*)d_in[5];
  const float* pbins = (const float*)d_in[37];
  const float* ebins = (const float*)d_in[38];
  const float* pemb  = (const float*)d_in[39];
  const float* eemb  = (const float*)d_in[40];

  auto P = [&](int base, int off) { return (const float*)d_in[base + off]; };
  // per-predictor param offsets: 0:w1 1:b1 2:g1 3:bt1 4:w2 5:b2 6:g2 7:bt2 8:lw 9:lb
  const int DUR = 7, PIT = 17, ENE = 27;

  // ---- workspace carve-up ----
  size_t o = 0;
  auto carve = [&](size_t bytes) { size_t r = o; o += (bytes + 255) & ~(size_t)255; return r; };
  char* ws = (char*)d_ws;
  unsigned short* x_bf   = (unsigned short*)(ws + carve((size_t)B_ * S_ * D_ * 2));
  unsigned short* xr_bf  = (unsigned short*)(ws + carve((size_t)B_ * T_ * D_ * 2));
  unsigned short* hA     = (unsigned short*)(ws + carve((size_t)B_ * T_ * F_ * 2));
  unsigned short* hB     = (unsigned short*)(ws + carve((size_t)B_ * T_ * F_ * 2));
  unsigned short* wpack  = (unsigned short*)(ws + carve((size_t)6 * 3 * 256 * 256 * 2));
  int*            cum    = (int*)(ws + carve((size_t)B_ * S_ * 4));
  int*            idx    = (int*)(ws + carve((size_t)B_ * T_ * 4));
  int*            mel    = (int*)(ws + carve((size_t)B_ * 4));

  const size_t per_layer = (size_t)3 * 256 * 256;
  float* out = (float*)d_out;
  float* out_main = out;                                  // [B,T,D]
  float* out_ldur = out_main + (size_t)B_ * T_ * D_;      // [B,S]
  float* out_pit  = out_ldur + (size_t)B_ * S_;           // [B,T]
  float* out_ene  = out_pit  + (size_t)B_ * T_;           // [B,T]
  float* out_len  = out_ene  + (size_t)B_ * T_;           // [B]
  float* out_mask = out_len  + B_;                        // [B,T]

  // 1) weights -> packed bf16 fragments (layers: durW1 durW2 pitW1 pitW2 eneW1 eneW2)
  pack_weights_kernel<<<(6 * 3 * 256 * 256 + 255) / 256, 256, 0, stream>>>(
      P(DUR, 0), P(DUR, 4), P(PIT, 0), P(PIT, 4), P(ENE, 0), P(ENE, 4), wpack);
  // 2) x -> bf16
  convert_x_kernel<<<(B_ * S_ * D_ + 255) / 256, 256, 0, stream>>>(x, x_bf, B_ * S_ * D_);
  // 3) duration cumsum + mel_len
  cumsum_kernel<<<B_, S_, 0, stream>>>(dtg, cum, mel);
  // 4) length-regulator gather indices
  lr_index_kernel<<<(B_ * T_ + 255) / 256, 256, 0, stream>>>(cum, idx);
  // 5) expanded activations (bf16)
  gather_xr_kernel<<<B_ * T_, 256, 0, stream>>>(x, idx, mel, xr_bf);

  // 6) duration predictor on x [B,S]
  conv_ln_kernel<<<B_ * (S_ / 16), 256, 0, stream>>>(x_bf, wpack + 0 * per_layer,
      P(DUR, 1), P(DUR, 2), P(DUR, 3), hA, S_);
  conv_ln_kernel<<<B_ * (S_ / 16), 256, 0, stream>>>(hA, wpack + 1 * per_layer,
      P(DUR, 5), P(DUR, 6), P(DUR, 7), hB, S_);
  linear_kernel<<<B_ * S_ / 8, 256, 0, stream>>>(hB, P(DUR, 8), P(DUR, 9), mel,
      out_ldur, S_, 0);

  // 7) pitch predictor on xr [B,T]
  conv_ln_kernel<<<B_ * (T_ / 16), 256, 0, stream>>>(xr_bf, wpack + 2 * per_layer,
      P(PIT, 1), P(PIT, 2), P(PIT, 3), hA, T_);
  conv_ln_kernel<<<B_ * (T_ / 16), 256, 0, stream>>>(hA, wpack + 3 * per_layer,
      P(PIT, 5), P(PIT, 6), P(PIT, 7), hB, T_);
  linear_kernel<<<B_ * T_ / 8, 256, 0, stream>>>(hB, P(PIT, 8), P(PIT, 9), mel,
      out_pit, T_, 1);

  // 8) energy predictor on xr [B,T]
  conv_ln_kernel<<<B_ * (T_ / 16), 256, 0, stream>>>(xr_bf, wpack + 4 * per_layer,
      P(ENE, 1), P(ENE, 2), P(ENE, 3), hA, T_);
  conv_ln_kernel<<<B_ * (T_ / 16), 256, 0, stream>>>(hA, wpack + 5 * per_layer,
      P(ENE, 5), P(ENE, 6), P(ENE, 7), hB, T_);
  linear_kernel<<<B_ * T_ / 8, 256, 0, stream>>>(hB, P(ENE, 8), P(ENE, 9), mel,
      out_ene, T_, 1);

  // 9) out = xr + pitch_emb + energy_emb  (f32 re-gather, bucketize, add)
  final_out_kernel<<<B_ * T_, 256, 0, stream>>>(x, idx, mel, ptg, etg, pbins, ebins,
      pemb, eemb, out_main);
  // 10) mel_len + mel_mask
  mask_len_kernel<<<(B_ * T_ + 255) / 256, 256, 0, stream>>>(mel, out_len, out_mask);
}
